// FiLMContextLinearModel_66760971649632
// MI455X (gfx1250) — matrix operand order repaired and checked by reading, not compiled, not run on hardware
//
#include <hip/hip_runtime.h>

// ---------------- CDNA5 WMMA types ----------------
typedef __attribute__((ext_vector_type(16))) __bf16 v16bf;
typedef __attribute__((ext_vector_type(8)))  float  v8f;

union FragB16 { v16bf v; unsigned int u[8]; };

__device__ __forceinline__ unsigned short f2bf(float f) {
  unsigned int x = __float_as_uint(f);
  x = x + 0x7FFFu + ((x >> 16) & 1u);   // round-to-nearest-even
  return (unsigned short)(x >> 16);
}
__device__ __forceinline__ float gelu_exact(float x) {
  return 0.5f * x * (1.0f + erff(x * 0.70710678118654752440f));
}
__device__ __forceinline__ float wred(float v, int width) {
  for (int off = width >> 1; off > 0; off >>= 1) v += __shfl_xor(v, off, width);
  return v;
}

// ---------------- problem constants ----------------
constexpr long long NT = 16384;      // B*T tokens
constexpr int Tdim = 4096;

constexpr long long algn(long long x) { return (x + 255) & ~255LL; }
constexpr long long OFF_HBF = 0;                                   // h bf16 16384x512
constexpr long long OFF_ZBF = algn(OFF_HBF + NT*512*2);            // z bf16
constexpr long long OFF_ZF  = algn(OFF_ZBF + NT*512*2);            // z f32
constexpr long long OFF_FIN = algn(OFF_ZF  + NT*512*4);            // film_in bf16 16384x640
constexpr long long OFF_GB  = algn(OFF_FIN + NT*640*2);            // gamma|beta f32 16384x1024
constexpr long long OFF_BL  = algn(OFF_GB  + NT*1024*4);           // bl f32
constexpr long long OFF_BS  = algn(OFF_BL  + NT*4);                // b_soft f32
constexpr long long OFF_ST  = algn(OFF_BS  + NT*4);                // per-token (mu,rstd) of h
constexpr long long OFF_V   = algn(OFF_ST  + NT*8);                // v[src2][head4][NT] f32
constexpr long long OFF_PO  = algn(OFF_V   + 8*NT*4);              // pooled bf16 4x16384x512
constexpr long long OFF_P1  = algn(OFF_PO  + 4*NT*512*2);          // p1 bf16 4x16384x512
constexpr long long OFF_WF  = algn(OFF_P1  + 4*NT*512*2);          // film_w^T bf16 1024x640
constexpr long long OFF_WA1 = algn(OFF_WF  + 1024LL*640*2);        // attn_w1^T bf16 4x256x512
constexpr long long OFF_WP1 = algn(OFF_WA1 + 4LL*256*512*2);       // proj_w1^T bf16 4x512x512
constexpr long long OFF_WQ  = algn(OFF_WP1 + 4LL*512*512*2);       // 64x512
constexpr long long OFF_WR  = algn(OFF_WQ  + 64LL*512*2);          // 13x512
constexpr long long OFF_WB  = algn(OFF_WR  + 13LL*512*2);          // 13x512
constexpr long long OFF_WK  = algn(OFF_WB  + 13LL*512*2);          // 25x512
constexpr long long WS_NEED = algn(OFF_WK  + 25LL*512*2);

// ---------------- conversion kernels ----------------
__global__ void k_cvt(const float* __restrict__ in, unsigned short* __restrict__ out, long long n) {
  long long i = (long long)blockIdx.x * blockDim.x + threadIdx.x;
  if (i < n) out[i] = f2bf(in[i]);
}
// in: [z][K][N] f32 row-major -> out: [z][N][K] bf16 (transposed for WMMA B)
__global__ void k_cvt_t(const float* __restrict__ in, unsigned short* __restrict__ out, int K, int N) {
  long long base = (long long)blockIdx.z * K * N;
  long long i = (long long)blockIdx.x * blockDim.x + threadIdx.x;
  if (i < (long long)K * N) {
    int k = (int)(i / N), n = (int)(i % N);
    out[base + (long long)n * K + k] = f2bf(in[base + i]);
  }
}

// ---------------- bl + LN(h) stats (one wave per token) ----------------
__global__ void k_bl_stats(const float* __restrict__ h, const float* __restrict__ w_bnd,
                           const float* __restrict__ b_bnd, float* __restrict__ bl,
                           float* __restrict__ stats, float* __restrict__ out) {
  int wid = threadIdx.x >> 5, lane = threadIdx.x & 31;
  long long t = (long long)blockIdx.x * 8 + wid;
  const float* hr = h + t * 512;
  float dot = 0.f, s = 0.f, s2 = 0.f;
#pragma unroll
  for (int j = 0; j < 16; ++j) {
    int d = lane + 32 * j;
    float x = hr[d];
    dot += x * w_bnd[d]; s += x; s2 += x * x;
  }
  dot = wred(dot, 32); s = wred(s, 32); s2 = wred(s2, 32);
  if (lane == 0) {
    float blv = dot + b_bnd[0];
    bl[t] = blv;
    float mu = s * (1.f / 512.f);
    float var = s2 * (1.f / 512.f) - mu * mu;
    stats[2 * t] = mu;
    stats[2 * t + 1] = rsqrtf(var + 1e-5f);
    out[t * 116 + 115] = blv;            // last output column = bl
  }
}

// ---------------- conv(9) + sigmoid ----------------
__global__ void k_conv(const float* __restrict__ bl, const float* __restrict__ ck,
                       const float* __restrict__ cb, float* __restrict__ bs) {
  int idx = blockIdx.x * blockDim.x + threadIdx.x;   // NT
  int b = idx >> 12, tt = idx & (Tdim - 1);
  float acc = cb[0];
#pragma unroll
  for (int j = 0; j < 9; ++j) {
    int tn = tt + j - 4;
    if (tn >= 0 && tn < Tdim) acc += bl[(b << 12) + tn] * ck[j];
  }
  bs[idx] = 1.f / (1.f + __expf(-acc));
}

// ---------------- film_in = LN([h, eb]) -> bf16 (one wave per token) ----------------
__global__ void k_filmin(const float* __restrict__ h, const float* __restrict__ bs,
                         const float* __restrict__ e0, const float* __restrict__ e1,
                         const float* __restrict__ g, const float* __restrict__ bb,
                         unsigned short* __restrict__ fin) {
  int wid = threadIdx.x >> 5, lane = threadIdx.x & 31;
  long long t = (long long)blockIdx.x * 8 + wid;
  float bsv = bs[t];
  float x[20]; float s = 0.f, s2 = 0.f;
#pragma unroll
  for (int j = 0; j < 20; ++j) {
    int d = lane + 32 * j;
    float v = (d < 512) ? h[t * 512 + d]
                        : (bsv * e1[d - 512] + (1.f - bsv) * e0[d - 512]);
    x[j] = v; s += v; s2 += v * v;
  }
  s = wred(s, 32); s2 = wred(s2, 32);
  float mu = s * (1.f / 640.f);
  float rstd = rsqrtf(s2 * (1.f / 640.f) - mu * mu + 1e-5f);
#pragma unroll
  for (int j = 0; j < 20; ++j) {
    int d = lane + 32 * j;
    fin[t * 640 + d] = f2bf((x[j] - mu) * rstd * g[d] + bb[d]);
  }
}

// ---------------- tiled WMMA GEMM: C = A(bf16,MxK) @ B^T(bf16,NxK) + bias ----------------
// Block tile 128x128, wave tile 32x64 (2x4 WMMA frags, 8 wmma per K-step).
// EPI 0: store f32 ; EPI 1: gelu -> store bf16
template <int EPI>
__global__ __launch_bounds__(256) void k_gemm(
    const unsigned short* __restrict__ A, long long strideA, int lda,
    const unsigned short* __restrict__ Bt, long long strideB,
    const float* __restrict__ bias, long long strideBias,
    void* __restrict__ Out, long long strideOut, int ldo, int Ksz) {
  __shared__ __align__(16) unsigned short As[128][40];
  __shared__ __align__(16) unsigned short Bs[128][40];
  const int z = blockIdx.z;
  A += (long long)z * strideA; Bt += (long long)z * strideB; bias += (long long)z * strideBias;
  const int tid = threadIdx.x, wid = tid >> 5, lane = tid & 31;
  const int wrow = wid & 3, wcol = wid >> 2;          // 4 row groups x 2 col groups
  const long long row0 = (long long)blockIdx.x * 128;
  const int col0 = blockIdx.y * 128;
  const int ldrow = tid >> 2, ldseg = tid & 3;        // 64 rows per pass, 2 passes
  const int m = lane & 15, half = lane >> 4;
  v8f acc[2][4] = {};
  for (int k0 = 0; k0 < Ksz; k0 += 32) {
    const unsigned short* ga0 = A + (row0 + ldrow) * lda + k0 + ldseg * 8;
    const unsigned short* ga1 = A + (row0 + 64 + ldrow) * lda + k0 + ldseg * 8;
    const unsigned short* gb0 = Bt + (long long)(col0 + ldrow) * Ksz + k0 + ldseg * 8;
    const unsigned short* gb1 = Bt + (long long)(col0 + 64 + ldrow) * Ksz + k0 + ldseg * 8;
    uint4 av0 = *(const uint4*)(const void*)ga0;
    uint4 av1 = *(const uint4*)(const void*)ga1;
    uint4 bv0 = *(const uint4*)(const void*)gb0;
    uint4 bv1 = *(const uint4*)(const void*)gb1;
    if (k0 + 32 < Ksz) {
      __builtin_prefetch((const void*)(ga0 + 32), 0, 1);  // global_prefetch_b8
      __builtin_prefetch((const void*)(gb0 + 32), 0, 1);
    }
    __syncthreads();
    *(uint4*)(void*)&As[ldrow][ldseg * 8]      = av0;
    *(uint4*)(void*)&As[64 + ldrow][ldseg * 8] = av1;
    *(uint4*)(void*)&Bs[ldrow][ldseg * 8]      = bv0;
    *(uint4*)(void*)&Bs[64 + ldrow][ldseg * 8] = bv1;
    __syncthreads();
    FragB16 fa[2];
#pragma unroll
    for (int r = 0; r < 2; ++r)
#pragma unroll
      for (int j = 0; j < 8; ++j) {
        int ka = 2 * j + 16 * half;   // A frag layout
        fa[r].u[j] = *(const unsigned int*)(const void*)&As[wrow * 32 + r * 16 + m][ka];
      }
#pragma unroll
    for (int f = 0; f < 4; ++f) {
      FragB16 fb;
#pragma unroll
      for (int j = 0; j < 8; ++j) {
        int kb = 2 * j + 16 * half;                                  // B frag layout
        fb.u[j] = *(const unsigned int*)(const void*)&Bs[wcol * 64 + f * 16 + m][kb];
      }
      acc[0][f] = __builtin_amdgcn_wmma_f32_16x16x32_bf16(false, fa[0].v, false, fb.v, (short)0, acc[0][f], false, false);
      acc[1][f] = __builtin_amdgcn_wmma_f32_16x16x32_bf16(false, fa[1].v, false, fb.v, (short)0, acc[1][f], false, false);
    }
  }
#pragma unroll
  for (int r = 0; r < 2; ++r)
#pragma unroll
    for (int f = 0; f < 4; ++f) {
      int c = col0 + wcol * 64 + f * 16 + m;
      float bvv = bias[c];
#pragma unroll
      for (int i = 0; i < 8; ++i) {
        long long rr = row0 + wrow * 32 + r * 16 + i + 8 * half;
        float v0 = acc[r][f][i] + bvv;
        if (EPI == 0) {
          float* O = (float*)Out + (long long)z * strideOut;
          O[rr * ldo + c] = v0;
        } else {
          unsigned short* O = (unsigned short*)Out + (long long)z * strideOut;
          O[rr * ldo + c] = f2bf(gelu_exact(v0));
        }
      }
    }
}

// ---------------- z = LN(h)*(1+gamma)+beta ----------------
__global__ void k_z(const float* __restrict__ h, const float* __restrict__ stats,
                    const float* __restrict__ gb, const float* __restrict__ g,
                    const float* __restrict__ bb, float* __restrict__ zf,
                    unsigned short* __restrict__ zb) {
  long long i = (long long)blockIdx.x * blockDim.x + threadIdx.x;  // NT*512
  long long t = i >> 9; int d = (int)(i & 511);
  float mu = stats[2 * t], rstd = stats[2 * t + 1];
  float hn = (h[i] - mu) * rstd * g[d] + bb[d];
  float zv = hn * (1.f + gb[t * 1024 + d]) + gb[t * 1024 + 512 + d];
  zf[i] = zv; zb[i] = f2bf(zv);
}

// ---------------- fused: v[src,head,t] = gelu(X @ W1 + b1) . w2 + b2 ----------------
// X = z (src 0) or h (src 1); W1^T bf16 [head][256][512]; full N=256 per block
__global__ __launch_bounds__(256) void k_uv(
    const unsigned short* __restrict__ hbf, const unsigned short* __restrict__ zbf,
    const unsigned short* __restrict__ wa1t, const float* __restrict__ b1,
    const float* __restrict__ w2, const float* __restrict__ b2, float* __restrict__ vout) {
  __shared__ __align__(16) unsigned short As[64][40];
  __shared__ __align__(16) unsigned short Bs[256][40];
  __shared__ float vacc[64];
  const int idx8 = blockIdx.y, head = idx8 & 3, src = idx8 >> 2;
  const unsigned short* X = src ? hbf : zbf;
  const unsigned short* Bt = wa1t + (long long)head * 256 * 512;
  const long long row0 = (long long)blockIdx.x * 64;
  const int tid = threadIdx.x, wid = tid >> 5, lane = tid & 31;
  const int wrow = wid & 3, wcol = wid >> 2;
  const int ldrow = tid >> 2, ldseg = tid & 3;
  const int nl = lane & 15, half = lane >> 4;
  if (tid < 64) vacc[tid] = 0.f;
  v8f acc[8] = {};
  for (int k0 = 0; k0 < 512; k0 += 32) {
    const unsigned short* ga = X + (row0 + ldrow) * 512 + k0 + ldseg * 8;
    uint4 av = *(const uint4*)(const void*)ga;
    uint4 bv[4];
#pragma unroll
    for (int r = 0; r < 4; ++r)
      bv[r] = *(const uint4*)(const void*)(Bt + (long long)(ldrow + 64 * r) * 512 + k0 + ldseg * 8);
    if (k0 + 32 < 512) __builtin_prefetch((const void*)(ga + 32), 0, 1);
    __syncthreads();
    *(uint4*)(void*)&As[ldrow][ldseg * 8] = av;
#pragma unroll
    for (int r = 0; r < 4; ++r) *(uint4*)(void*)&Bs[ldrow + 64 * r][ldseg * 8] = bv[r];
    __syncthreads();
    FragB16 fa;
#pragma unroll
    for (int j = 0; j < 8; ++j) {
      int ka = 2 * j + 16 * half;
      fa.u[j] = *(const unsigned int*)(const void*)&As[wrow * 16 + nl][ka];
    }
#pragma unroll
    for (int f = 0; f < 8; ++f) {
      FragB16 fb;
#pragma unroll
      for (int j = 0; j < 8; ++j) {
        int kb = 2 * j + 16 * half;
        fb.u[j] = *(const unsigned int*)(const void*)&Bs[wcol * 128 + f * 16 + nl][kb];
      }
      acc[f] = __builtin_amdgcn_wmma_f32_16x16x32_bf16(false, fa.v, false, fb.v, (short)0, acc[f], false, false);
    }
  }
  float part[8] = {0.f, 0.f, 0.f, 0.f, 0.f, 0.f, 0.f, 0.f};
#pragma unroll
  for (int f = 0; f < 8; ++f) {
    int n = wcol * 128 + f * 16 + nl;
    float b1v = b1[head * 256 + n], w2v = w2[head * 256 + n];
#pragma unroll
    for (int i = 0; i < 8; ++i) part[i] += gelu_exact(acc[f][i] + b1v) * w2v;
  }
#pragma unroll
  for (int i = 0; i < 8; ++i) part[i] = wred(part[i], 16);
  if (nl == 0) {
#pragma unroll
    for (int i = 0; i < 8; ++i) atomicAdd(&vacc[wrow * 16 + 8 * half + i], part[i]);
  }
  __syncthreads();
  if (tid < 64) vout[(long long)idx8 * NT + row0 + tid] = vacc[tid] + b2[head];
}

// ---------------- softmax over 10 channels + pooling (one wave per (head,t)) ----------------
__global__ void k_pool(const float* __restrict__ v, const float* __restrict__ zf,
                       const float* __restrict__ h, unsigned short* __restrict__ pooled) {
  int wid = threadIdx.x >> 5, lane = threadIdx.x & 31;
  long long w = (long long)blockIdx.x * 8 + wid;         // 0..4*NT-1
  int head = (int)(w >> 14); long long t = w & (NT - 1);
  int b = (int)(t >> 12), tt = (int)(t & (Tdim - 1));
  const float* vz = v + (long long)head * NT;
  const float* vl = v + (long long)(4 + head) * NT;
  float sc[10]; int nidx[9];
  sc[0] = vz[t];
#pragma unroll
  for (int j = 0; j < 9; ++j) {
    int tn = tt + j - 4; tn = tn < 0 ? 0 : (tn > Tdim - 1 ? Tdim - 1 : tn);
    nidx[j] = tn; sc[j + 1] = vl[(long long)(b << 12) + tn];
  }
  float mx = sc[0];
#pragma unroll
  for (int i = 1; i < 10; ++i) mx = fmaxf(mx, sc[i]);
  float den = 0.f;
#pragma unroll
  for (int i = 0; i < 10; ++i) { sc[i] = __expf(sc[i] - mx); den += sc[i]; }
  float inv = 1.f / den;
#pragma unroll
  for (int e = 0; e < 16; ++e) {
    int d = lane + 32 * e;
    float a = sc[0] * inv * zf[t * 512 + d];
#pragma unroll
    for (int j = 0; j < 9; ++j)
      a += sc[j + 1] * inv * h[((long long)(b << 12) + nidx[j]) * 512 + d];
    pooled[((long long)head * NT + t) * 512 + d] = f2bf(a);
  }
}

// ---------------- vocab projections (direct-global WMMA, one wave per 16x16 tile) ----------------
__global__ __launch_bounds__(256) void k_vocab(
    const unsigned short* __restrict__ p1,
    const unsigned short* __restrict__ wq, const unsigned short* __restrict__ wr,
    const unsigned short* __restrict__ wb, const unsigned short* __restrict__ wk,
    const float* __restrict__ bq, const float* __restrict__ br,
    const float* __restrict__ bbv, const float* __restrict__ bk,
    float* __restrict__ out) {
  const int headArr[8]  = {0, 0, 0, 0, 1, 2, 3, 3};
  const int vidArr[8]   = {0, 0, 0, 0, 1, 2, 3, 3};
  const int vbaseArr[8] = {0, 16, 32, 48, 0, 0, 0, 16};
  const int widthArr[8] = {64, 64, 64, 64, 13, 13, 25, 25};
  const int outcArr[8]  = {0, 16, 32, 48, 64, 77, 90, 106};
  const int jt = blockIdx.y;
  const unsigned short* wt; const float* bias;
  switch (vidArr[jt]) {
    case 0:  wt = wq; bias = bq; break;
    case 1:  wt = wr; bias = br; break;
    case 2:  wt = wb; bias = bbv; break;
    default: wt = wk; bias = bk; break;
  }
  const int head = headArr[jt], width = widthArr[jt], vbase = vbaseArr[jt], outc = outcArr[jt];
  const int wid = threadIdx.x >> 5, lane = threadIdx.x & 31;
  const long long row0 = ((long long)blockIdx.x * 8 + wid) * 16;
  const unsigned short* Ab = p1 + (long long)head * NT * 512;
  const int m = lane & 15, half = lane >> 4;
  const int nc = vbase + m;
  const int ncc = nc < width ? nc : (width - 1);
  v8f acc = {};
  for (int k0 = 0; k0 < 512; k0 += 32) {
    FragB16 fa, fb;
#pragma unroll
    for (int j = 0; j < 8; ++j) {
      int ka = 2 * j + 16 * half;
      fa.u[j] = *(const unsigned int*)(const void*)(Ab + (row0 + m) * 512 + k0 + ka);
      int kb = 2 * j + 16 * half;
      fb.u[j] = *(const unsigned int*)(const void*)(wt + (long long)ncc * 512 + k0 + kb);
    }
    acc = __builtin_amdgcn_wmma_f32_16x16x32_bf16(false, fa.v, false, fb.v, (short)0, acc, false, false);
  }
  if (nc < width) {
    float bv = bias[nc];
#pragma unroll
    for (int i = 0; i < 8; ++i) {
      long long r = row0 + i + 8 * half;
      out[r * 116 + outc + m] = acc[i] + bv;
    }
  }
}

// ---------------- launch ----------------
extern "C" void kernel_launch(void* const* d_in, const int* in_sizes, int n_in,
                              void* d_out, int out_size, void* d_ws, size_t ws_size,
                              hipStream_t stream) {
  (void)in_sizes; (void)n_in; (void)out_size;
  if (ws_size < (size_t)WS_NEED) return;
  const float* h       = (const float*)d_in[0];
  const float* w_bnd   = (const float*)d_in[1];
  const float* b_bnd   = (const float*)d_in[2];
  const float* conv_k  = (const float*)d_in[3];
  const float* conv_b  = (const float*)d_in[4];
  const float* e0      = (const float*)d_in[5];
  const float* e1      = (const float*)d_in[6];
  const float* ln_in_g = (const float*)d_in[7];
  const float* ln_in_b = (const float*)d_in[8];
  const float* ln_h_g  = (const float*)d_in[9];
  const float* ln_h_b  = (const float*)d_in[10];
  const float* film_w  = (const float*)d_in[11];
  const float* film_b  = (const float*)d_in[12];
  const float* attn_w1 = (const float*)d_in[13];
  const float* attn_b1 = (const float*)d_in[14];
  const float* attn_w2 = (const float*)d_in[15];
  const float* attn_b2 = (const float*)d_in[16];
  const float* proj_w1 = (const float*)d_in[17];
  const float* proj_b1 = (const float*)d_in[18];
  const float* w2q = (const float*)d_in[19]; const float* b2q = (const float*)d_in[20];
  const float* w2r = (const float*)d_in[21]; const float* b2r = (const float*)d_in[22];
  const float* w2b = (const float*)d_in[23]; const float* b2b = (const float*)d_in[24];
  const float* w2k = (const float*)d_in[25]; const float* b2k = (const float*)d_in[26];
  float* out = (float*)d_out;

  char* ws = (char*)d_ws;
  unsigned short* hbf  = (unsigned short*)(ws + OFF_HBF);
  unsigned short* zbf  = (unsigned short*)(ws + OFF_ZBF);
  float*          zf   = (float*)(ws + OFF_ZF);
  unsigned short* fin  = (unsigned short*)(ws + OFF_FIN);
  float*          gb   = (float*)(ws + OFF_GB);
  float*          bl   = (float*)(ws + OFF_BL);
  float*          bs   = (float*)(ws + OFF_BS);
  float*          st   = (float*)(ws + OFF_ST);
  float*          vv   = (float*)(ws + OFF_V);
  unsigned short* po   = (unsigned short*)(ws + OFF_PO);
  unsigned short* p1   = (unsigned short*)(ws + OFF_P1);
  unsigned short* wft  = (unsigned short*)(ws + OFF_WF);
  unsigned short* wa1t = (unsigned short*)(ws + OFF_WA1);
  unsigned short* wp1t = (unsigned short*)(ws + OFF_WP1);
  unsigned short* wqt  = (unsigned short*)(ws + OFF_WQ);
  unsigned short* wrt  = (unsigned short*)(ws + OFF_WR);
  unsigned short* wbt  = (unsigned short*)(ws + OFF_WB);
  unsigned short* wkt  = (unsigned short*)(ws + OFF_WK);

  // weight / activation conversions
  k_cvt<<<(int)((NT * 512 + 255) / 256), 256, 0, stream>>>(h, hbf, NT * 512);
  k_cvt_t<<<dim3((640 * 1024 + 255) / 256, 1, 1), 256, 0, stream>>>(film_w, wft, 640, 1024);
  k_cvt_t<<<dim3((512 * 256 + 255) / 256, 1, 4), 256, 0, stream>>>(attn_w1, wa1t, 512, 256);
  k_cvt_t<<<dim3((512 * 512 + 255) / 256, 1, 4), 256, 0, stream>>>(proj_w1, wp1t, 512, 512);
  k_cvt_t<<<dim3((512 * 64 + 255) / 256, 1, 1), 256, 0, stream>>>(w2q, wqt, 512, 64);
  k_cvt_t<<<dim3((512 * 13 + 255) / 256, 1, 1), 256, 0, stream>>>(w2r, wrt, 512, 13);
  k_cvt_t<<<dim3((512 * 13 + 255) / 256, 1, 1), 256, 0, stream>>>(w2b, wbt, 512, 13);
  k_cvt_t<<<dim3((512 * 25 + 255) / 256, 1, 1), 256, 0, stream>>>(w2k, wkt, 512, 25);

  // boundary logit + LN(h) stats, conv+sigmoid, film_in LN
  k_bl_stats<<<(int)(NT / 8), 256, 0, stream>>>(h, w_bnd, b_bnd, bl, st, out);
  k_conv<<<(int)(NT / 256), 256, 0, stream>>>(bl, conv_k, conv_b, bs);
  k_filmin<<<(int)(NT / 8), 256, 0, stream>>>(h, bs, e0, e1, ln_in_g, ln_in_b, fin);

  // film GEMM: 16384x640 @ 640x1024 -> gamma|beta  (128x128 block tiles)
  k_gemm<0><<<dim3((int)(NT / 128), 1024 / 128, 1), 256, 0, stream>>>(
      fin, 0, 640, wft, 0, film_b, 0, (void*)gb, 0, 1024, 640);

  // z
  k_z<<<(int)(NT * 512 / 256), 256, 0, stream>>>(h, st, gb, ln_h_g, ln_h_b, zf, zbf);

  // fused u/v: scores precompute for z-channel (src 0) and local rows (src 1)
  k_uv<<<dim3((int)(NT / 64), 8, 1), 256, 0, stream>>>(hbf, zbf, wa1t, attn_b1, attn_w2, attn_b2, vv);

  // softmax + pooling
  k_pool<<<(int)(4 * NT / 8), 256, 0, stream>>>(vv, zf, h, po);

  // per-head proj GEMM with gelu epilogue: 4 x (16384x512 @ 512x512), 128x128 tiles
  k_gemm<1><<<dim3((int)(NT / 128), 512 / 128, 4), 256, 0, stream>>>(
      po, NT * 512, 512, wp1t, 512LL * 512, proj_b1, 512, (void*)p1, NT * 512, 512, 512);

  // vocab projections into the 116-wide output
  k_vocab<<<dim3((int)(NT / 128), 8, 1), 256, 0, stream>>>(
      p1, wqt, wrt, wbt, wkt, b2q, b2r, b2b, b2k, out);
}
